// RBFLayer_71313636983006
// MI455X (gfx1250) — matrix operand order repaired and checked by reading, not compiled, not run on hardware
//
#include <hip/hip_runtime.h>
#include <math.h>

typedef __attribute__((ext_vector_type(16))) _Float16 v16h;
typedef __attribute__((ext_vector_type(8)))  _Float16 v8h;
typedef __attribute__((ext_vector_type(8)))  float    v8f;
typedef __attribute__((ext_vector_type(4)))  float    v4f;

#define UNITS      512
#define DIMS       64
#define ROWPAD     72          // halfs per padded center row (144B stride -> conflict-free b128 B-frag loads)
#define BLOCK_ROWS 128         // 8 waves x 16 rows
#define EPSV       1e-12f

__global__ __launch_bounds__(256, 1) void rbf_wmma_kernel(
    const float* __restrict__ x,
    const float* __restrict__ centers,
    const float* __restrict__ gamma_raw,
    float* __restrict__ out)
{
    __shared__ __align__(16) _Float16 lds_b[UNITS * ROWPAD]; // 73728 B: all centers as f16
    __shared__ float lds_cn[UNITS];                          // ||c||^2
    __shared__ float lds_g [UNITS];                          // softplus(gamma_raw)
    __shared__ float lds_xn[BLOCK_ROWS];                     // ||x||^2 + eps per block row

    const int tid = threadIdx.x;
    const long long m0 = (long long)blockIdx.x * BLOCK_ROWS;

    // ---- Stage all centers into LDS as f16 (B-matrix source), c_norm, gamma ----
    for (int u = tid; u < UNITS; u += 256) {
        const float* cr = centers + u * DIMS;
        v8h* dst = (v8h*)&lds_b[u * ROWPAD];
        float s = 0.f;
#pragma unroll
        for (int k = 0; k < DIMS; k += 8) {
            v4f a = *(const v4f*)(cr + k);
            v4f b = *(const v4f*)(cr + k + 4);
            s = fmaf(a.x, a.x, s); s = fmaf(a.y, a.y, s);
            s = fmaf(a.z, a.z, s); s = fmaf(a.w, a.w, s);
            s = fmaf(b.x, b.x, s); s = fmaf(b.y, b.y, s);
            s = fmaf(b.z, b.z, s); s = fmaf(b.w, b.w, s);
            v8h h;
            h[0] = (_Float16)a.x; h[1] = (_Float16)a.y;
            h[2] = (_Float16)a.z; h[3] = (_Float16)a.w;
            h[4] = (_Float16)b.x; h[5] = (_Float16)b.y;
            h[6] = (_Float16)b.z; h[7] = (_Float16)b.w;
            dst[k >> 3] = h;
        }
        lds_cn[u] = s;
        lds_g[u]  = log1pf(__expf(gamma_raw[u]));   // softplus
    }

    // ---- ||x_m||^2 + eps for this block's 128 rows ----
    if (tid < BLOCK_ROWS) {
        const float* xr = x + (m0 + tid) * DIMS;
        float s = 0.f;
#pragma unroll
        for (int k = 0; k < DIMS; k += 4) {
            v4f v = *(const v4f*)(xr + k);
            s = fmaf(v.x, v.x, s); s = fmaf(v.y, v.y, s);
            s = fmaf(v.z, v.z, s); s = fmaf(v.w, v.w, s);
        }
        lds_xn[tid] = s + EPSV;
    }
    __syncthreads();

    // ---- per-wave: 16 rows of x against all 512 units ----
    const int wave = tid >> 5;
    const int lane = tid & 31;
    const int hi   = lane >> 4;          // half-wave select
    const int lo   = lane & 15;

    // Build the two A fragments (K = 0..31 and 32..63), ISA 16-bit 16x32 A layout:
    // lane (hi,lo): row M = lo; VGPR j<4 holds K = 2j + 8*hi; j>=4 holds K = 16 + 2(j-4) + 8*hi.
    const float* xr = x + (m0 + wave * 16 + lo) * DIMS;
    v16h af[2];
#pragma unroll
    for (int w = 0; w < 2; ++w) {
        const float* p  = xr + 32 * w + 8 * hi;
        v4f c0 = *(const v4f*)(p);
        v4f c1 = *(const v4f*)(p + 4);
        v4f c2 = *(const v4f*)(p + 16);
        v4f c3 = *(const v4f*)(p + 20);
        v16h a;
        a[0]  = (_Float16)c0.x; a[1]  = (_Float16)c0.y; a[2]  = (_Float16)c0.z; a[3]  = (_Float16)c0.w;
        a[4]  = (_Float16)c1.x; a[5]  = (_Float16)c1.y; a[6]  = (_Float16)c1.z; a[7]  = (_Float16)c1.w;
        a[8]  = (_Float16)c2.x; a[9]  = (_Float16)c2.y; a[10] = (_Float16)c2.z; a[11] = (_Float16)c2.w;
        a[12] = (_Float16)c3.x; a[13] = (_Float16)c3.y; a[14] = (_Float16)c3.z; a[15] = (_Float16)c3.w;
        af[w] = a;
    }

    // Row norms for this lane's 8 output rows (C/D layout: M = r + 8*hi)
    float xn[8];
#pragma unroll
    for (int r = 0; r < 8; ++r)
        xn[r] = lds_xn[wave * 16 + 8 * hi + r];

    float* outp = out + (m0 + (long long)(wave * 16 + 8 * hi)) * (long long)UNITS + lo;

    for (int nt = 0; nt < (UNITS / 16); ++nt) {
        // B fragment, ISA 16-bit 32x16 B layout: column N = lo, lane holds K = 8*hi..+7 and 16+8*hi..+7.
        const _Float16* bp = &lds_b[(unsigned)(nt * 16 + lo) * ROWPAD + 8 * hi];
        v8h b0a = *(const v8h*)(bp);
        v8h b0b = *(const v8h*)(bp + 16);
        v8h b1a = *(const v8h*)(bp + 32);   // + K-offset 32 halfs
        v8h b1b = *(const v8h*)(bp + 48);
        v16h b0, b1;
#pragma unroll
        for (int i = 0; i < 8; ++i) {
            b0[i] = b0a[i]; b0[i + 8] = b0b[i];
            b1[i] = b1a[i]; b1[i + 8] = b1b[i];
        }

        v8f acc = {};
        acc = __builtin_amdgcn_wmma_f32_16x16x32_f16(false, af[0], false, b0,
                                                     (short)0, acc, false, false);
        acc = __builtin_amdgcn_wmma_f32_16x16x32_f16(false, af[1], false, b1,
                                                     (short)0, acc, false, false);

        const float g  = lds_g [nt * 16 + lo];
        const float cn = lds_cn[nt * 16 + lo];
#pragma unroll
        for (int r = 0; r < 8; ++r) {
            float d2 = fmaxf(fmaf(-2.0f, acc[r], xn[r] + cn), 0.0f);
            outp[(long long)r * UNITS + nt * 16] = __expf(-g * d2);
        }
    }
}

extern "C" void kernel_launch(void* const* d_in, const int* in_sizes, int n_in,
                              void* d_out, int out_size, void* d_ws, size_t ws_size,
                              hipStream_t stream) {
    const float* x         = (const float*)d_in[0];
    const float* centers   = (const float*)d_in[1];
    const float* gamma_raw = (const float*)d_in[2];
    float* out             = (float*)d_out;

    const int batch = in_sizes[0] / DIMS;          // 131072
    const int grid  = batch / BLOCK_ROWS;          // 1024 blocks of 256 threads (8 wave32)
    rbf_wmma_kernel<<<grid, 256, 0, stream>>>(x, centers, gamma_raw, out);
}
